// MultiHeadedAttention_5497558139710
// MI455X (gfx1250) — matrix operand order repaired and checked by reading, not compiled, not run on hardware
//
#include <hip/hip_runtime.h>
#include <hip/hip_bf16.h>
#include <stdint.h>

// ---------------------------------------------------------------------------
// MultiHeadedAttention for MI455X (gfx1250), wave32 + v_wmma_f32_16x16x32_f16
// B=2, S=2048, D=1024, H=16, HW=64
//
// Pipeline:
//   0. cvt_kernel        : f32 -> f16 one-time conversion of X / W operands
//   1. proj_kernel  (x3) : per-head GEMMs -> Q[b,h,s,w], K[b,h,t,w], V[b,h,w,t]
//                          (32 rows x 64 cols per wave: 8 WMMA / 12 b128 loads)
//   2. attn_kernel       : flash attention (online softmax), -> ab[b,s,h,w]
//   3. outproj_kernel    : collapse GEMM (M=4096,N=1024,K=1024) + bias -> f32
// ---------------------------------------------------------------------------

#define B_  2
#define S_  2048
#define D_  1024
#define H_  16
#define HW_ 64
#define MASK_FILL_ -100000000.0f

typedef __attribute__((ext_vector_type(16))) _Float16 v16h;
typedef __attribute__((ext_vector_type(8)))  _Float16 v8h;
typedef __attribute__((ext_vector_type(8)))  float    v8f;
typedef __attribute__((ext_vector_type(4)))  float    v4f;

union H16 { v16h v; v8h h[2]; };

static __device__ __forceinline__ v8f wmma_f16(v16h a, v16h b, v8f c) {
    // D = A(16x32 f16) * B(32x16 f16) + C(16x16 f32)
    return __builtin_amdgcn_wmma_f32_16x16x32_f16(
        /*neg_a=*/false, a, /*neg_b=*/false, b,
        /*c_mod=*/(short)0, c, /*reuse_a=*/false, /*reuse_b=*/false);
}

// ---------------------------------------------------------------------------
// Kernel 0: one-time f32 -> f16 conversion (8 elements / thread, b128 loads)
// ---------------------------------------------------------------------------
__global__ __launch_bounds__(256) void cvt_kernel(
    const float* __restrict__ src, _Float16* __restrict__ dst, int n8)
{
    const int i = blockIdx.x * blockDim.x + threadIdx.x;
    if (i >= n8) return;
    const float* p = src + (size_t)i * 8;
    v4f a = *(const v4f*)p;
    v4f b = *(const v4f*)(p + 4);
    v8h r;
    r[0] = (_Float16)a[0]; r[1] = (_Float16)a[1];
    r[2] = (_Float16)a[2]; r[3] = (_Float16)a[3];
    r[4] = (_Float16)b[0]; r[5] = (_Float16)b[1];
    r[6] = (_Float16)b[2]; r[7] = (_Float16)b[3];
    *(v8h*)(dst + (size_t)i * 8) = r;
}

// ---------------------------------------------------------------------------
// Kernel 1: per-head projection GEMM (all-f16), 32x64 tile per wave.
//   X: [B,S,D] f16, W: [H,HW,D] f16, bias: [H,HW] f32 -> dst f16
// K loop over D in steps of 32: 2 A-frag tiles + 4 B-frags -> 8 WMMAs.
// ---------------------------------------------------------------------------
__global__ __launch_bounds__(64) void proj_kernel(
    const _Float16* __restrict__ X, const _Float16* __restrict__ W,
    const float* __restrict__ bias, _Float16* __restrict__ dst,
    int transposeOut)
{
    const int lane = threadIdx.x & 31;
    const int widx = threadIdx.x >> 5;
    const int l16  = lane & 15;
    const int half = lane >> 4;
    const int s0   = (blockIdx.x * 2 + widx) * 32;   // 32 rows per wave
    const int b    = blockIdx.y / H_;
    const int h    = blockIdx.y % H_;

    const v8f vzero = {0.f,0.f,0.f,0.f,0.f,0.f,0.f,0.f};
    v8f acc[2][4] = { { vzero, vzero, vzero, vzero },
                      { vzero, vzero, vzero, vzero } };

    const int akoff = half ? 16 : 0;   // A frag: K split {0..15}/{16..31}
    const int bkoff = half ? 16 : 0;   // B frag: K split {0..15}/{16..31}
    const _Float16* arow[2] = {
        X + ((size_t)b * S_ + (s0 + l16)) * D_,
        X + ((size_t)b * S_ + (s0 + 16 + l16)) * D_ };
    const _Float16* brow0 = W + ((size_t)(h * HW_ + l16)) * D_ + bkoff;

    // prologue: fragments for kk = 0
    H16 ac[2], bc[4];
#pragma unroll
    for (int mt = 0; mt < 2; ++mt) {
        ac[mt].h[0] = *(const v8h*)(arow[mt] + akoff);
        ac[mt].h[1] = *(const v8h*)(arow[mt] + 8 + akoff);
    }
#pragma unroll
    for (int nt = 0; nt < 4; ++nt) {
        const _Float16* br = brow0 + (size_t)nt * 16 * D_;
        bc[nt].h[0] = *(const v8h*)br;
        bc[nt].h[1] = *(const v8h*)(br + 8);
    }

    for (int kk = 0; kk < D_; kk += 32) {
        const int kn = (kk + 32 < D_) ? kk + 32 : 0;   // wrap on last iter
        H16 an[2], bn[4];
#pragma unroll
        for (int mt = 0; mt < 2; ++mt) {
            an[mt].h[0] = *(const v8h*)(arow[mt] + kn + akoff);
            an[mt].h[1] = *(const v8h*)(arow[mt] + kn + 8 + akoff);
        }
#pragma unroll
        for (int nt = 0; nt < 4; ++nt) {
            const _Float16* br = brow0 + (size_t)nt * 16 * D_ + kn;
            bn[nt].h[0] = *(const v8h*)br;
            bn[nt].h[1] = *(const v8h*)(br + 8);
        }
#pragma unroll
        for (int mt = 0; mt < 2; ++mt)
#pragma unroll
            for (int nt = 0; nt < 4; ++nt)
                acc[mt][nt] = wmma_f16(ac[mt].v, bc[nt].v, acc[mt][nt]);
#pragma unroll
        for (int mt = 0; mt < 2; ++mt) ac[mt] = an[mt];
#pragma unroll
        for (int nt = 0; nt < 4; ++nt) bc[nt] = bn[nt];
    }

#pragma unroll
    for (int mt = 0; mt < 2; ++mt) {
#pragma unroll
        for (int nt = 0; nt < 4; ++nt) {
            const int w = nt * 16 + l16;
            const float bw = bias[h * HW_ + w];
            if (!transposeOut) {
                // dst[b,h,row,w] — scattered b16 stores
#pragma unroll
                for (int r = 0; r < 8; ++r) {
                    const int row = s0 + mt * 16 + r + 8 * half;
                    dst[(((size_t)b * H_ + h) * S_ + row) * HW_ + w] =
                        (_Float16)(acc[mt][nt][r] + bw);
                }
            } else {
                // dst[b,h,w,t]: 8 consecutive t per lane -> one b128 store
                v8h pk;
#pragma unroll
                for (int r = 0; r < 8; ++r)
                    pk[r] = (_Float16)(acc[mt][nt][r] + bw);
                *(v8h*)(dst + (((size_t)b * H_ + h) * HW_ + w) * S_ +
                        s0 + mt * 16 + 8 * half) = pk;
            }
        }
    }
}

// ---------------------------------------------------------------------------
// helper: load the 4 K-fragments (two 16-col t-subtiles x two 32-wide w-chunks)
// ---------------------------------------------------------------------------
static __device__ __forceinline__ void loadK4(
    const _Float16* __restrict__ kbase, int t, int l16, int bkoff, H16 kf[4])
{
#pragma unroll
    for (int ct = 0; ct < 2; ++ct) {
        const _Float16* krow = kbase + (size_t)(t + ct * 16 + l16) * HW_;
        kf[ct * 2 + 0].h[0] = *(const v8h*)(krow + bkoff);
        kf[ct * 2 + 0].h[1] = *(const v8h*)(krow + bkoff + 8);
        kf[ct * 2 + 1].h[0] = *(const v8h*)(krow + 32 + bkoff);
        kf[ct * 2 + 1].h[1] = *(const v8h*)(krow + 32 + bkoff + 8);
    }
}

// ---------------------------------------------------------------------------
// Kernel 2: flash attention per (b,h, 16-row tile), pipelined.
//   qb,kb: [b,h,s,w] f16;  vb: [b,h,w,t] f16 (transposed); mask: [S,S] u8
//   out ab: [b,s,h,w] f16  (attn / sqrt(HW))
// ---------------------------------------------------------------------------
__global__ __launch_bounds__(64) void attn_kernel(
    const _Float16* __restrict__ qb, const _Float16* __restrict__ kb,
    const _Float16* __restrict__ vb, const unsigned char* __restrict__ mask,
    _Float16* __restrict__ ab)
{
    __shared__ _Float16 sP[2][16 * 32];   // per-wave P re-layout buffer

    const int lane = threadIdx.x & 31;
    const int widx = threadIdx.x >> 5;
    const int l16  = lane & 15;
    const int half = lane >> 4;
    const int s0   = (blockIdx.x * 2 + widx) * 16;
    const int bh   = blockIdx.y;
    const int b    = bh / H_;
    const int h    = bh % H_;

    const int akoff = half ? 16 : 0;
    const int bkoff = half ? 16 : 0;

    // Q fragments for w-chunks [0,32) and [32,64): kept in registers all sweep
    H16 qa[2];
    const _Float16* qrow = qb + ((size_t)bh * S_ + s0 + l16) * HW_;
    qa[0].h[0] = *(const v8h*)(qrow + akoff);
    qa[0].h[1] = *(const v8h*)(qrow + 8 + akoff);
    qa[1].h[0] = *(const v8h*)(qrow + 32 + akoff);
    qa[1].h[1] = *(const v8h*)(qrow + 40 + akoff);

    const v8f vzero = {0.f,0.f,0.f,0.f,0.f,0.f,0.f,0.f};
    v8f acc[4] = { vzero, vzero, vzero, vzero };  // 16 x 64 f32 output accumulator
    float m[8], l[8];
#pragma unroll
    for (int r = 0; r < 8; ++r) { m[r] = -3.0e38f; l[r] = 0.0f; }

    const _Float16* kbase = kb + (size_t)bh * S_ * HW_;

    // prologue: K fragments for t = 0
    H16 kf[4];
    loadK4(kbase, 0, l16, bkoff, kf);

    for (int t = 0; t < S_; t += 32) {
        // ---- logits from K fragments loaded one iteration ago
        v8f lg[2];
        lg[0] = wmma_f16(qa[0].v, kf[0].v, vzero);
        lg[0] = wmma_f16(qa[1].v, kf[1].v, lg[0]);
        lg[1] = wmma_f16(qa[0].v, kf[2].v, vzero);
        lg[1] = wmma_f16(qa[1].v, kf[3].v, lg[1]);

        // ---- issue next K tile + this tile's V fragments now; they complete
        //      while the softmax VALU section below runs
        const int tn = (t + 32 < S_) ? t + 32 : 0;     // wrap on last iter
        H16 kn[4];
        loadK4(kbase, tn, l16, bkoff, kn);
        H16 vf[4];
#pragma unroll
        for (int nt = 0; nt < 4; ++nt) {
            const _Float16* vrow =
                vb + ((size_t)bh * HW_ + nt * 16 + l16) * S_ + t + bkoff;
            vf[nt].h[0] = *(const v8h*)vrow;
            vf[nt].h[1] = *(const v8h*)(vrow + 8);
        }

        // ---- mask fill (C layout: row = r + 8*half, col = l16)
#pragma unroll
        for (int ct = 0; ct < 2; ++ct) {
#pragma unroll
            for (int r = 0; r < 8; ++r) {
                const int row = s0 + r + 8 * half;
                const int col = t + ct * 16 + l16;
                if (mask[(size_t)row * S_ + col]) lg[ct][r] = MASK_FILL_;
            }
        }

        // ---- online softmax statistics (reductions across 16-lane halves)
        float sc[8];
#pragma unroll
        for (int r = 0; r < 8; ++r) {
            float tm = fmaxf(lg[0][r], lg[1][r]);
            tm = fmaxf(tm, __shfl_xor(tm, 1, 32));
            tm = fmaxf(tm, __shfl_xor(tm, 2, 32));
            tm = fmaxf(tm, __shfl_xor(tm, 4, 32));
            tm = fmaxf(tm, __shfl_xor(tm, 8, 32));
            const float mn = fmaxf(m[r], tm);
            sc[r] = __expf(m[r] - mn);
            m[r] = mn;
            const float p0 = __expf(lg[0][r] - mn);
            const float p1 = __expf(lg[1][r] - mn);
            lg[0][r] = p0; lg[1][r] = p1;
            float rs = p0 + p1;
            rs += __shfl_xor(rs, 1, 32);
            rs += __shfl_xor(rs, 2, 32);
            rs += __shfl_xor(rs, 4, 32);
            rs += __shfl_xor(rs, 8, 32);
            l[r] = l[r] * sc[r] + rs;
        }

        // ---- rescale running accumulator
#pragma unroll
        for (int nt = 0; nt < 4; ++nt)
#pragma unroll
            for (int r = 0; r < 8; ++r) acc[nt][r] *= sc[r];

        // ---- re-layout P (C fragment -> A fragment) through LDS
        __syncthreads();
#pragma unroll
        for (int ct = 0; ct < 2; ++ct)
#pragma unroll
            for (int r = 0; r < 8; ++r)
                sP[widx][(r + 8 * half) * 32 + ct * 16 + l16] = (_Float16)lg[ct][r];
        __syncthreads();

        H16 pf;
        pf.h[0] = *(const v8h*)&sP[widx][l16 * 32 + akoff];
        pf.h[1] = *(const v8h*)&sP[widx][l16 * 32 + 8 + akoff];

        // ---- acc += P(16x32) * V(32x64) with the prefetched V fragments
#pragma unroll
        for (int nt = 0; nt < 4; ++nt)
            acc[nt] = wmma_f16(pf.v, vf[nt].v, acc[nt]);

        // rotate K double-buffer
#pragma unroll
        for (int i = 0; i < 4; ++i) kf[i] = kn[i];
    }

    // ---- finalize: (1/l) * 1/sqrt(HW);  store [b, s, h, w] f16
    float inv[8];
#pragma unroll
    for (int r = 0; r < 8; ++r) inv[r] = 0.125f / l[r];
#pragma unroll
    for (int nt = 0; nt < 4; ++nt) {
        const int w = nt * 16 + l16;
#pragma unroll
        for (int r = 0; r < 8; ++r) {
            const int row = s0 + r + 8 * half;
            ab[(((size_t)b * S_ + row) * H_ + h) * HW_ + w] =
                (_Float16)(acc[nt][r] * inv[r]);
        }
    }
}

// ---------------------------------------------------------------------------
// Kernel 3: output projection GEMM, 32x64 tile per wave.
//   out[m,o] = ab[m,k] * Wo[o,k] + bo[o]
//   M = B*S = 4096, N = D = 1024, K = H*HW = 1024; Wo pre-converted f16.
// ---------------------------------------------------------------------------
__global__ __launch_bounds__(64) void outproj_kernel(
    const _Float16* __restrict__ A, const _Float16* __restrict__ Wo,
    const float* __restrict__ bo, float* __restrict__ out)
{
    const int lane = threadIdx.x & 31;
    const int widx = threadIdx.x >> 5;
    const int l16  = lane & 15;
    const int half = lane >> 4;
    const int m0   = (blockIdx.x * 2 + widx) * 32;   // 32 rows per wave
    const int n0   = blockIdx.y * 64;

    const v8f vzero = {0.f,0.f,0.f,0.f,0.f,0.f,0.f,0.f};
    v8f acc[2][4] = { { vzero, vzero, vzero, vzero },
                      { vzero, vzero, vzero, vzero } };

    const int akoff = half ? 16 : 0;
    const int bkoff = half ? 16 : 0;
    const _Float16* arow[2] = {
        A + (size_t)(m0 + l16) * (H_ * HW_),
        A + (size_t)(m0 + 16 + l16) * (H_ * HW_) };
    const _Float16* brow0 = Wo + (size_t)(n0 + l16) * (H_ * HW_) + bkoff;

    H16 ac[2], bc[4];
#pragma unroll
    for (int mt = 0; mt < 2; ++mt) {
        ac[mt].h[0] = *(const v8h*)(arow[mt] + akoff);
        ac[mt].h[1] = *(const v8h*)(arow[mt] + 8 + akoff);
    }
#pragma unroll
    for (int nt = 0; nt < 4; ++nt) {
        const _Float16* br = brow0 + (size_t)nt * 16 * (H_ * HW_);
        bc[nt].h[0] = *(const v8h*)br;
        bc[nt].h[1] = *(const v8h*)(br + 8);
    }

    for (int kk = 0; kk < H_ * HW_; kk += 32) {
        const int kn = (kk + 32 < H_ * HW_) ? kk + 32 : 0;
        H16 an[2], bn[4];
#pragma unroll
        for (int mt = 0; mt < 2; ++mt) {
            an[mt].h[0] = *(const v8h*)(arow[mt] + kn + akoff);
            an[mt].h[1] = *(const v8h*)(arow[mt] + kn + 8 + akoff);
        }
#pragma unroll
        for (int nt = 0; nt < 4; ++nt) {
            const _Float16* br = brow0 + (size_t)nt * 16 * (H_ * HW_) + kn;
            bn[nt].h[0] = *(const v8h*)br;
            bn[nt].h[1] = *(const v8h*)(br + 8);
        }
#pragma unroll
        for (int mt = 0; mt < 2; ++mt)
#pragma unroll
            for (int nt = 0; nt < 4; ++nt)
                acc[mt][nt] = wmma_f16(ac[mt].v, bc[nt].v, acc[mt][nt]);
#pragma unroll
        for (int mt = 0; mt < 2; ++mt) ac[mt] = an[mt];
#pragma unroll
        for (int nt = 0; nt < 4; ++nt) bc[nt] = bn[nt];
    }

#pragma unroll
    for (int mt = 0; mt < 2; ++mt) {
#pragma unroll
        for (int nt = 0; nt < 4; ++nt) {
            const int n = n0 + nt * 16 + l16;
            const float bias = bo[n];
#pragma unroll
            for (int r = 0; r < 8; ++r) {
                const int row = m0 + mt * 16 + r + 8 * half;
                out[(size_t)row * D_ + n] = acc[mt][nt][r] + bias;
            }
        }
    }
}

// ---------------------------------------------------------------------------
// Launcher
// in order: query, key, value, mask, Wq, bq, Wk, bk, Wv, bv, Wo, bo
// ---------------------------------------------------------------------------
extern "C" void kernel_launch(void* const* d_in, const int* in_sizes, int n_in,
                              void* d_out, int out_size, void* d_ws, size_t ws_size,
                              hipStream_t stream) {
    const float* query = (const float*)d_in[0];
    const float* key   = (const float*)d_in[1];
    const float* value = (const float*)d_in[2];
    const unsigned char* mask = (const unsigned char*)d_in[3];
    const float* Wq = (const float*)d_in[4];
    const float* bq = (const float*)d_in[5];
    const float* Wk = (const float*)d_in[6];
    const float* bk = (const float*)d_in[7];
    const float* Wv = (const float*)d_in[8];
    const float* bv = (const float*)d_in[9];
    const float* Wo = (const float*)d_in[10];
    const float* bo = (const float*)d_in[11];
    float* out = (float*)d_out;

    // f16 workspace layout (element counts)
    const size_t NX = (size_t)B_ * S_ * D_;        // 4,194,304 (x3 activations)
    const size_t NW = (size_t)H_ * HW_ * D_;       // 1,048,576 (x4 weights)
    const size_t NQ = (size_t)B_ * H_ * S_ * HW_;  // 4,194,304 (x4 intermediates)
    _Float16* xq = (_Float16*)d_ws;
    _Float16* xk = xq + NX;
    _Float16* xv = xk + NX;
    _Float16* wq = xv + NX;
    _Float16* wk = wq + NW;
    _Float16* wv = wk + NW;
    _Float16* wo = wv + NW;
    _Float16* qb = wo + NW;
    _Float16* kb = qb + NQ;
    _Float16* vb = kb + NQ;
    _Float16* ab = vb + NQ;

    // ---- 0) one-time f32 -> f16 conversions
    {
        const int nx8 = (int)(NX / 8), nw8 = (int)(NW / 8);
        const dim3 cb(256, 1, 1);
        cvt_kernel<<<dim3((nx8 + 255) / 256), cb, 0, stream>>>(query, xq, nx8);
        cvt_kernel<<<dim3((nx8 + 255) / 256), cb, 0, stream>>>(key,   xk, nx8);
        cvt_kernel<<<dim3((nx8 + 255) / 256), cb, 0, stream>>>(value, xv, nx8);
        cvt_kernel<<<dim3((nw8 + 255) / 256), cb, 0, stream>>>(Wq, wq, nw8);
        cvt_kernel<<<dim3((nw8 + 255) / 256), cb, 0, stream>>>(Wk, wk, nw8);
        cvt_kernel<<<dim3((nw8 + 255) / 256), cb, 0, stream>>>(Wv, wv, nw8);
        cvt_kernel<<<dim3((nw8 + 255) / 256), cb, 0, stream>>>(Wo, wo, nw8);
    }

    const dim3 blk(64, 1, 1);                 // 2 waves per block
    const dim3 gProj(S_ / 64, B_ * H_, 1);    // 32-row tile per wave
    proj_kernel<<<gProj, blk, 0, stream>>>(xq, wq, bq, qb, 0);
    proj_kernel<<<gProj, blk, 0, stream>>>(xk, wk, bk, kb, 0);
    proj_kernel<<<gProj, blk, 0, stream>>>(xv, wv, bv, vb, 1);  // transposed

    const dim3 gAttn(S_ / 32, B_ * H_, 1);    // 16-row tile per wave
    attn_kernel<<<gAttn, blk, 0, stream>>>(qb, kb, vb, mask, ab);

    const dim3 gOut((B_ * S_) / 64, D_ / 64, 1);  // 32-row tile per wave
    outproj_kernel<<<gOut, blk, 0, stream>>>(ab, wo, bo, out);
}
